// Model_58128087384364
// MI455X (gfx1250) — matrix operand order repaired and checked by reading, not compiled, not run on hardware
//
#include <hip/hip_runtime.h>
#include <hip/hip_bf16.h>
#include <math.h>

typedef __bf16 bf16_t;
typedef __attribute__((ext_vector_type(16))) __bf16 v16bf;
typedef __attribute__((ext_vector_type(8)))  float  v8f;

#define DEV __device__ __forceinline__

// ---------------- dims ----------------
#define NB 32
#define OH1 254
#define OW1 65
#define OH2 125
#define OW2 17
#define DENC 544
#define TENC 125
#define HID 128
#define GATE 384
#define TDEC 127
#define ODIM 4096

DEV v8f wmma_bf16(v16bf a, v16bf b, v8f c) {
  return __builtin_amdgcn_wmma_f32_16x16x32_bf16(false, a, false, b, (short)0, c,
                                                 false, false);
}
DEV float sigm(float x) { return 1.f / (1.f + __expf(-x)); }

// A fragment element map (16x32 bf16): lane -> row m = lane&15,
// k's = {kb..kb+7, kb+16..kb+23}, kb = (lane>=16)*8.  Load from a per-lane
// row pointer p (already offset by m*ldk + kb).
DEV v16bf load_a_ptr(const bf16_t* __restrict__ p) {
  v16bf a;
#pragma unroll
  for (int e = 0; e < 8; ++e) a[e] = p[e];
#pragma unroll
  for (int e = 0; e < 8; ++e) a[8 + e] = p[16 + e];
  return a;
}
DEV v16bf load_a_rm(const bf16_t* __restrict__ A, int ldk) {
  const int lane = threadIdx.x & 31;
  return load_a_ptr(A + (size_t)(lane & 15) * ldk + ((lane >> 4) << 3));
}
// B fragment from packed layout: pk[(chunk*32+lane)*16 + e], 32B/lane contiguous
DEV v16bf load_b_packed(const bf16_t* __restrict__ pk, int chunk) {
  const int lane = threadIdx.x & 31;
  return *(const v16bf*)(pk + ((size_t)chunk * 32 + lane) * 16);
}
// C/D: VGPR v -> M = v + (lane>=16?8:0), N = lane&15
template <typename F> DEV void store_c_frag(v8f c, F f) {
  const int lane = threadIdx.x & 31;
  const int n = lane & 15;
  const int mb = (lane >> 4) << 3;
#pragma unroll
  for (int v = 0; v < 8; ++v) f(mb + v, n, c[v]);
}

// ---------------- pack W(N,K) row-major f32 -> B-fragment bf16 ----------------
__global__ void k_pack_b(const float* __restrict__ w, bf16_t* __restrict__ dst,
                         int N, int K) {
  int idx = blockIdx.x * 256 + threadIdx.x;
  if (idx >= N * K) return;
  int e = idx & 15;
  int lane = (idx >> 4) & 31;
  int chunk = idx >> 9;
  int KC = K >> 5;
  int nt = chunk / KC, kc = chunk % KC;
  int n = nt * 16 + (lane & 15);
  int k = kc * 32 + ((lane >> 4) << 3) + (e & 7) + ((e >> 3) << 4);
  dst[idx] = (bf16_t)w[(size_t)n * K + k];
}

__global__ void k_cvt_bf16(const float* __restrict__ src, bf16_t* __restrict__ dst,
                           int n) {
  int i = blockIdx.x * 256 + threadIdx.x;
  if (i < n) dst[i] = (bf16_t)src[i];
}

// ---------------- conv1: implicit GEMM, M=32*254*65, N=32, K=5x32 ----------------
__global__ void k_conv1(const bf16_t* __restrict__ x, const bf16_t* __restrict__ wpk,
                        const float* __restrict__ bias, bf16_t* __restrict__ out) {
  int job = blockIdx.x * 8 + (threadIdx.x >> 5);
  if (job >= 33020 * 2) return;
  int mt = job >> 1, nt = job & 1;
  int m0 = mt * 16, n0 = nt * 16;
  const int lane = threadIdx.x & 31;
  // per-lane A row decode (once)
  {
    int ma = m0 + (lane & 15);
    int ab = ma / (OH1 * OW1), ar = ma % (OH1 * OW1);
    int aoh = ar / OW1, aow = ar % OW1;
    const bf16_t* arow =
        x + ((size_t)ab * 512 + aoh * 2) * 161 + aow * 2 + ((lane >> 4) << 3);
    v8f acc = {};
#pragma unroll
    for (int kh = 0; kh < 5; ++kh)
      acc = wmma_bf16(load_a_ptr(arow + kh * 161), load_b_packed(wpk, nt * 5 + kh), acc);
    // C store: decode first row once, carry-increment the rest
    int mg = m0 + ((lane >> 4) << 3);
    int b = mg / (OH1 * OW1), r = mg % (OH1 * OW1);
    int oh = r / OW1, ow = r % OW1;
    int n = lane & 15;
    float bv = bias[n0 + n];
#pragma unroll
    for (int v = 0; v < 8; ++v) {
      float o = fmaxf(acc[v] + bv, 0.f);
      out[(((size_t)b * 32 + (n0 + n)) * OH1 + oh) * OW1 + ow] = (bf16_t)o;
      if (++ow == OW1) { ow = 0; if (++oh == OH1) { oh = 0; ++b; } }
    }
  }
}

// ---------------- conv2: implicit GEMM, M=32*125*17, N=32, K=160x32 ----------------
__global__ void k_conv2(const bf16_t* __restrict__ in, const bf16_t* __restrict__ wpk,
                        const float* __restrict__ bias, bf16_t* __restrict__ out) {
  int job = blockIdx.x * 8 + (threadIdx.x >> 5);
  if (job >= 4250 * 2) return;
  int mt = job >> 1, nt = job & 1;
  int m0 = mt * 16, n0 = nt * 16;
  const int lane = threadIdx.x & 31;
  // per-lane A row decode (once); offset excludes ci,kh terms
  int ma = m0 + (lane & 15);
  int ab = ma / (OH2 * OW2), ar = ma % (OH2 * OW2);
  int aoh = ar / OW2, aow = ar % OW2;
  const bf16_t* abase = in + (size_t)ab * (32 * OH1 * OW1) +
                        (size_t)(aoh * 2) * OW1 + aow * 2 + ((lane >> 4) << 3);
  v8f acc = {};
  for (int ci = 0; ci < 32; ++ci) {
    const bf16_t* ap = abase + (size_t)ci * (OH1 * OW1);
#pragma unroll
    for (int kh = 0; kh < 5; ++kh)
      acc = wmma_bf16(load_a_ptr(ap + kh * OW1),
                      load_b_packed(wpk, nt * 160 + ci * 5 + kh), acc);
  }
  // C store to (b, t, f*c): decode once + carry-increment
  int mg = m0 + ((lane >> 4) << 3);
  int b = mg / (OH2 * OW2), r = mg % (OH2 * OW2);
  int oh = r / OW2, ow = r % OW2;
  int n = lane & 15;
  float bv = bias[n0 + n];
#pragma unroll
  for (int v = 0; v < 8; ++v) {
    float o = fmaxf(acc[v] + bv, 0.f);
    out[((size_t)(b * TENC + oh)) * DENC + ow * 32 + (n0 + n)] = (bf16_t)o;
    if (++ow == OW2) { ow = 0; if (++oh == OH2) { oh = 0; ++b; } }
  }
}

// ---------------- encoder xg = h_enc @ w_ih^T + b_ih : M=4000,K=544,N=384 ----------------
__global__ void k_xg_enc(const bf16_t* __restrict__ henc, const bf16_t* __restrict__ wpk,
                         const float* __restrict__ bih, float* __restrict__ xg) {
  int job = blockIdx.x * 8 + (threadIdx.x >> 5);
  if (job >= 250 * 24) return;
  int mt = job / 24, nt = job % 24;
  int m0 = mt * 16, n0 = nt * 16;
  v8f acc = {};
  for (int kc = 0; kc < 17; ++kc) {
    v16bf af = load_a_rm(henc + (size_t)m0 * DENC + kc * 32, DENC);
    acc = wmma_bf16(af, load_b_packed(wpk, nt * 17 + kc), acc);
  }
  store_c_frag(acc, [&](int m, int n, float v) {
    xg[(size_t)(m0 + m) * GATE + n0 + n] = v + bih[n0 + n];
  });
}

// ---------------- encoder GRU scan: persistent, 1 block x 768 thr (24 waves) ------------
#define ENC_OFF_WHH 0
#define ENC_OFF_BHH (ENC_OFF_WHH + GATE * HID * 2)
#define ENC_OFF_HBF (ENC_OFF_BHH + GATE * 4)
#define ENC_OFF_HF  (ENC_OFF_HBF + NB * HID * 2)
#define ENC_OFF_HG  (ENC_OFF_HF + NB * HID * 4)
#define ENC_SMEM    (ENC_OFF_HG + NB * GATE * 4)

DEV void pack_whh_lds(bf16_t* s_whh, const float* __restrict__ whh) {
  for (int idx = threadIdx.x; idx < GATE * HID; idx += 768) {
    int e = idx & 15;
    int lane = (idx >> 4) & 31;
    int chunk = idx >> 9;  // nt*4 + kc, nt<24
    int nt = chunk >> 2, kc = chunk & 3;
    int n = nt * 16 + (lane & 15);
    int k = kc * 32 + ((lane >> 4) << 3) + (e & 7) + ((e >> 3) << 4);
    s_whh[idx] = (bf16_t)whh[(size_t)n * HID + k];
  }
}

// prefetch one 32-float cacheline per thread of next step's gate rows
DEV void prefetch_gates(const float* __restrict__ g, int T, int t) {
  const int tid = threadIdx.x;
  if (tid < 384 && t < T) {
    int b = tid / 12, ln = tid % 12;
    __builtin_prefetch(g + ((size_t)(b * T + t)) * GATE + ln * 32, 0, 1);
  }
}

// hg(32x384) = h(32x128) @ whh^T using 24 waves; reads s_hbf, writes s_hg
DEV void gru_matmul(const bf16_t* s_hbf, const bf16_t* s_whh, float* s_hg) {
  const int wave = threadIdx.x >> 5;
#pragma unroll
  for (int mt = 0; mt < 2; ++mt) {
    v8f acc = {};
#pragma unroll
    for (int kc = 0; kc < 4; ++kc) {
      v16bf af = load_a_rm(s_hbf + mt * 16 * HID + kc * 32, HID);
      acc = wmma_bf16(af, load_b_packed(s_whh, wave * 4 + kc), acc);
    }
    store_c_frag(acc, [&](int m, int n, float v) {
      s_hg[(mt * 16 + m) * GATE + wave * 16 + n] = v;
    });
  }
}

__global__ void k_enc_scan(const float* __restrict__ xg, const float* __restrict__ whh,
                           const float* __restrict__ bhh, float* __restrict__ eh) {
  extern __shared__ char smem[];
  bf16_t* s_whh = (bf16_t*)(smem + ENC_OFF_WHH);
  float*  s_bhh = (float*)(smem + ENC_OFF_BHH);
  bf16_t* s_hbf = (bf16_t*)(smem + ENC_OFF_HBF);
  float*  s_hf  = (float*)(smem + ENC_OFF_HF);
  float*  s_hg  = (float*)(smem + ENC_OFF_HG);
  const int tid = threadIdx.x;
  pack_whh_lds(s_whh, whh);
  for (int i = tid; i < GATE; i += 768) s_bhh[i] = bhh[i];
  for (int i = tid; i < NB * HID; i += 768) { s_hf[i] = 0.f; s_hbf[i] = (bf16_t)0.f; }
  prefetch_gates(xg, TENC, 0);
  __syncthreads();
  for (int t = 0; t < TENC; ++t) {
    prefetch_gates(xg, TENC, t + 1);  // hide next row's L2 latency behind WMMA
    gru_matmul(s_hbf, s_whh, s_hg);
    __syncthreads();
    for (int idx = tid; idx < NB * HID; idx += 768) {
      int b = idx >> 7, j = idx & 127;
      const float* xp = xg + ((size_t)(b * TENC + t)) * GATE;
      float xr = xp[j], xz = xp[HID + j], xn = xp[2 * HID + j];
      float hr = s_hg[b * GATE + j] + s_bhh[j];
      float hz = s_hg[b * GATE + HID + j] + s_bhh[HID + j];
      float hn = s_hg[b * GATE + 2 * HID + j] + s_bhh[2 * HID + j];
      float r = sigm(xr + hr), z = sigm(xz + hz);
      float n = tanhf(xn + r * hn);
      float h = (1.f - z) * n + z * s_hf[idx];
      s_hf[idx] = h;
      s_hbf[idx] = (bf16_t)h;
      eh[((size_t)(b * TENC + t)) * HID + j] = h;
    }
    __syncthreads();
  }
}

// ---------------- decoder xg precompute: M=32*127, K=32, N=384 ----------------
__global__ void k_xg_dec(const int* __restrict__ y, const float* __restrict__ emb,
                         const bf16_t* __restrict__ wpk, const float* __restrict__ bih,
                         float* __restrict__ xgd) {
  int job = blockIdx.x * 8 + (threadIdx.x >> 5);
  if (job >= 254 * 24) return;
  int mt = job / 24, nt = job % 24;
  int m0 = mt * 16, n0 = nt * 16;
  const int lane = threadIdx.x & 31;
  // per-lane embedding row (decode once)
  int mg = m0 + (lane & 15);
  int b = mg / TDEC, t = mg % TDEC;
  int tok = y[b * 128 + t];  // y[:, :-1]
  const float* erow = emb + (size_t)tok * 32 + ((lane >> 4) << 3);
  v16bf af;
#pragma unroll
  for (int e = 0; e < 8; ++e) af[e] = (bf16_t)erow[e];
#pragma unroll
  for (int e = 0; e < 8; ++e) af[8 + e] = (bf16_t)erow[16 + e];
  v8f acc = {};
  acc = wmma_bf16(af, load_b_packed(wpk, nt), acc);
  store_c_frag(acc, [&](int m, int n, float v) {
    xgd[(size_t)(m0 + m) * GATE + n0 + n] = v + bih[n0 + n];
  });
}

// ---------------- decoder scan: persistent, 1 block x 768 thr ----------------
#define DEC_OFF_SC  ENC_SMEM
#define DEC_SMEM    (DEC_OFF_SC + NB * TENC * 4)

DEV void attend_update(const float* __restrict__ eh, float* s_sc, float* s_hf,
                       bf16_t* s_hbf, bf16_t* __restrict__ out_bf, int pos) {
  const int tid = threadIdx.x;
  for (int idx = tid; idx < NB * TENC; idx += 768) {
    int b = idx / TENC, t = idx % TENC;
    const float* ep = eh + ((size_t)(b * TENC + t)) * HID;
    float s = 0.f;
#pragma unroll 4
    for (int j = 0; j < HID; ++j) s += ep[j] * s_hf[b * HID + j];
    s_sc[idx] = s;
  }
  __syncthreads();
  if (tid < NB) {
    float mx = -1e30f;
    for (int t = 0; t < TENC; ++t) mx = fmaxf(mx, s_sc[tid * TENC + t]);
    float sum = 0.f;
    for (int t = 0; t < TENC; ++t) {
      float e = __expf(s_sc[tid * TENC + t] - mx);
      s_sc[tid * TENC + t] = e;
      sum += e;
    }
    float inv = 1.f / sum;
    for (int t = 0; t < TENC; ++t) s_sc[tid * TENC + t] *= inv;
  }
  __syncthreads();
  for (int idx = tid; idx < NB * HID; idx += 768) {
    int b = idx >> 7, j = idx & 127;
    float c = 0.f;
#pragma unroll 4
    for (int t = 0; t < TENC; ++t)
      c += eh[((size_t)(b * TENC + t)) * HID + j] * s_sc[b * TENC + t];
    float v = s_hf[idx] + c;
    s_hf[idx] = v;
    s_hbf[idx] = (bf16_t)v;
    out_bf[((size_t)(b * 128 + pos)) * HID + j] = (bf16_t)v;
  }
  __syncthreads();
}

__global__ void k_dec_scan(const float* __restrict__ xgd, const float* __restrict__ whh,
                           const float* __restrict__ bhh, const float* __restrict__ eh,
                           const float* __restrict__ h_init, bf16_t* __restrict__ out_bf) {
  extern __shared__ char smem[];
  bf16_t* s_whh = (bf16_t*)(smem + ENC_OFF_WHH);
  float*  s_bhh = (float*)(smem + ENC_OFF_BHH);
  bf16_t* s_hbf = (bf16_t*)(smem + ENC_OFF_HBF);
  float*  s_hf  = (float*)(smem + ENC_OFF_HF);
  float*  s_hg  = (float*)(smem + ENC_OFF_HG);
  float*  s_sc  = (float*)(smem + DEC_OFF_SC);
  const int tid = threadIdx.x;
  pack_whh_lds(s_whh, whh);
  for (int i = tid; i < GATE; i += 768) s_bhh[i] = bhh[i];
  for (int i = tid; i < NB * HID; i += 768) {
    float v = h_init[i & 127];
    s_hf[i] = v;
    s_hbf[i] = (bf16_t)v;
  }
  prefetch_gates(xgd, TDEC, 0);
  __syncthreads();
  attend_update(eh, s_sc, s_hf, s_hbf, out_bf, 0);
  for (int t = 0; t < TDEC; ++t) {
    prefetch_gates(xgd, TDEC, t + 1);
    gru_matmul(s_hbf, s_whh, s_hg);
    __syncthreads();
    for (int idx = tid; idx < NB * HID; idx += 768) {
      int b = idx >> 7, j = idx & 127;
      const float* xp = xgd + ((size_t)(b * TDEC + t)) * GATE;
      float xr = xp[j], xz = xp[HID + j], xn = xp[2 * HID + j];
      float hr = s_hg[b * GATE + j] + s_bhh[j];
      float hz = s_hg[b * GATE + HID + j] + s_bhh[HID + j];
      float hn = s_hg[b * GATE + 2 * HID + j] + s_bhh[2 * HID + j];
      float r = sigm(xr + hr), z = sigm(xz + hz);
      float n = tanhf(xn + r * hn);
      float h = (1.f - z) * n + z * s_hf[idx];
      s_hf[idx] = h;
      s_hbf[idx] = (bf16_t)h;
    }
    __syncthreads();
    attend_update(eh, s_sc, s_hf, s_hbf, out_bf, t + 1);
  }
}

// ---------------- final FC: M=4096, N=4096, K=128; wave does 16x64 tile ----------------
__global__ void k_fc(const bf16_t* __restrict__ out_bf, const bf16_t* __restrict__ wpk,
                     const float* __restrict__ fcb, float* __restrict__ logits) {
  int job = blockIdx.x * 8 + (threadIdx.x >> 5);
  if (job >= 256 * 64) return;
  int mt = job >> 6, ng = job & 63;
  int m0 = mt * 16, n0 = ng * 64;
  v16bf af[4];
#pragma unroll
  for (int kc = 0; kc < 4; ++kc)
    af[kc] = load_a_rm(out_bf + (size_t)m0 * HID + kc * 32, HID);
#pragma unroll
  for (int nn = 0; nn < 4; ++nn) {
    int nt = (n0 >> 4) + nn;
    v8f acc = {};
#pragma unroll
    for (int kc = 0; kc < 4; ++kc)
      acc = wmma_bf16(af[kc], load_b_packed(wpk, nt * 4 + kc), acc);
    store_c_frag(acc, [&](int m, int n, float v) {
      logits[(size_t)(m0 + m) * ODIM + nt * 16 + n] = v + fcb[nt * 16 + n];
    });
  }
}

// ---------------- host ----------------
extern "C" void kernel_launch(void* const* d_in, const int* in_sizes, int n_in,
                              void* d_out, int out_size, void* d_ws, size_t ws_size,
                              hipStream_t stream) {
  const float* x        = (const float*)d_in[0];
  const int*   y        = (const int*)d_in[1];
  const float* conv1_w  = (const float*)d_in[2];
  const float* conv1_b  = (const float*)d_in[3];
  const float* conv2_w  = (const float*)d_in[4];
  const float* conv2_b  = (const float*)d_in[5];
  const float* rnn_w_ih = (const float*)d_in[6];
  const float* rnn_w_hh = (const float*)d_in[7];
  const float* rnn_b_ih = (const float*)d_in[8];
  const float* rnn_b_hh = (const float*)d_in[9];
  const float* emb      = (const float*)d_in[10];
  const float* dec_w_ih = (const float*)d_in[11];
  const float* dec_w_hh = (const float*)d_in[12];
  const float* dec_b_ih = (const float*)d_in[13];
  const float* dec_b_hh = (const float*)d_in[14];
  const float* fc_w     = (const float*)d_in[15];
  const float* fc_b     = (const float*)d_in[16];
  const float* h_init   = (const float*)d_in[17];
  float* logits = (float*)d_out;

  char* ws = (char*)d_ws;
  size_t off = 0;
  auto alloc = [&](size_t bytes) -> void* {
    void* p = ws + off;
    off = (off + bytes + 255) & ~(size_t)255;
    return p;
  };
  bf16_t* x_bf      = (bf16_t*)alloc((size_t)NB * 512 * 161 * 2);        // 5.3 MB
  bf16_t* conv1_out = (bf16_t*)alloc((size_t)NB * 32 * OH1 * OW1 * 2);   // 33.8 MB
  bf16_t* h_enc     = (bf16_t*)alloc((size_t)NB * TENC * DENC * 2);      // 4.4 MB
  float*  xg_enc    = (float*)alloc((size_t)NB * TENC * GATE * 4);       // 6.1 MB
  float*  eh        = (float*)alloc((size_t)NB * TENC * HID * 4);        // 2.0 MB
  float*  xgd       = (float*)alloc((size_t)NB * TDEC * GATE * 4);       // 6.2 MB
  bf16_t* out_bf    = (bf16_t*)alloc((size_t)NB * 128 * HID * 2);        // 1.0 MB
  bf16_t* w1_pk     = (bf16_t*)alloc((size_t)32 * 160 * 2);
  bf16_t* w2_pk     = (bf16_t*)alloc((size_t)32 * 5120 * 2);
  bf16_t* wih_pk    = (bf16_t*)alloc((size_t)GATE * DENC * 2);
  bf16_t* dwih_pk   = (bf16_t*)alloc((size_t)GATE * 32 * 2);
  bf16_t* fcw_pk    = (bf16_t*)alloc((size_t)ODIM * HID * 2);

  // one-time converts / fragment packing
  k_cvt_bf16<<<(NB * 512 * 161) / 256, 256, 0, stream>>>(x, x_bf, NB * 512 * 161);
  k_pack_b<<<20, 256, 0, stream>>>(conv1_w, w1_pk, 32, 160);
  k_pack_b<<<640, 256, 0, stream>>>(conv2_w, w2_pk, 32, 5120);
  k_pack_b<<<816, 256, 0, stream>>>(rnn_w_ih, wih_pk, GATE, DENC);
  k_pack_b<<<48, 256, 0, stream>>>(dec_w_ih, dwih_pk, GATE, 32);
  k_pack_b<<<2048, 256, 0, stream>>>(fc_w, fcw_pk, ODIM, HID);

  // pipeline
  k_conv1<<<8255, 256, 0, stream>>>(x_bf, w1_pk, conv1_b, conv1_out);
  k_conv2<<<1063, 256, 0, stream>>>(conv1_out, w2_pk, conv2_b, h_enc);
  k_xg_enc<<<750, 256, 0, stream>>>(h_enc, wih_pk, rnn_b_ih, xg_enc);
  k_enc_scan<<<1, 768, ENC_SMEM, stream>>>(xg_enc, rnn_w_hh, rnn_b_hh, eh);
  k_xg_dec<<<762, 256, 0, stream>>>(y, emb, dwih_pk, dec_b_ih, xgd);
  k_dec_scan<<<1, 768, DEC_SMEM, stream>>>(xgd, dec_w_hh, dec_b_hh, eh, h_init, out_bf);
  k_fc<<<2048, 256, 0, stream>>>(out_bf, fcw_pk, fc_b, logits);
}
